// TransformerBlock_47236050321946
// MI455X (gfx1250) — matrix operand order repaired and checked by reading, not compiled, not run on hardware
//
#include <hip/hip_runtime.h>
#include <hip/hip_bf16.h>
#include <math.h>

// ---------------------------------------------------------------------------
// Types for CDNA5 WMMA (wave32): A/B = 16 x bf16 per lane, C/D = 8 x f32.
// ---------------------------------------------------------------------------
typedef __attribute__((ext_vector_type(16))) __bf16 v16bf;
typedef __attribute__((ext_vector_type(8)))  float  v8f;

union FragU { uint4 q[2]; v16bf v; };

__device__ inline v8f vzero8() { v8f z; for (int e = 0; e < 8; ++e) z[e] = 0.f; return z; }

__device__ inline v8f wmma_bf16(const v16bf& a, const v16bf& b, const v8f& c) {
    return __builtin_amdgcn_wmma_f32_16x16x32_bf16(false, a, false, b, (short)0, c,
                                                   false, false);
}

// DPP16 row_ror butterfly reduction within each 16-lane half (pure VALU, no LDS).
// ctrl = 0x120 | n  ->  row_ror:n
template <int CTRL>
__device__ inline float dppf(float v) {
    return __builtin_bit_cast(
        float, __builtin_amdgcn_update_dpp(0, __builtin_bit_cast(int, v), CTRL, 0xf, 0xf,
                                           true));
}
__device__ inline float redmax16(float v) {
    v = fmaxf(v, dppf<0x121>(v));
    v = fmaxf(v, dppf<0x122>(v));
    v = fmaxf(v, dppf<0x124>(v));
    v = fmaxf(v, dppf<0x128>(v));
    return v;
}
__device__ inline float redsum16(float v) {
    v += dppf<0x121>(v);
    v += dppf<0x122>(v);
    v += dppf<0x124>(v);
    v += dppf<0x128>(v);
    return v;
}

#define Dm 1024
#define Sm 2048
#define Hh 16
#define DH 64
#define FF 4096
#define NTOK 4096  // B*S

// ---------------------------------------------------------------------------
// Weight transpose + f32 -> bf16 :  WT[n*K + k] = W[k*N + n]
// ---------------------------------------------------------------------------
__global__ __launch_bounds__(256) void wt_kernel(const float* __restrict__ W,
                                                 __bf16* __restrict__ WT,
                                                 int K, int N) {
    int idx = blockIdx.x * 256 + threadIdx.x;
    if (idx >= K * N) return;
    int k = idx / N, n = idx - k * N;
    WT[(size_t)n * K + k] = (__bf16)W[idx];
}

// ---------------------------------------------------------------------------
// V [tok, D] bf16  ->  Vt [b, h, dh(64), s(2048)] bf16  (keys contiguous)
// ---------------------------------------------------------------------------
__global__ __launch_bounds__(256) void vt_kernel(const __bf16* __restrict__ V,
                                                 __bf16* __restrict__ Vt) {
    int idx = blockIdx.x * 256 + threadIdx.x;   // over 4096*1024
    int row = idx >> 10;                        // b*S + s
    int col = idx & 1023;                       // h*64 + dh
    int b = row >> 11, s = row & 2047;
    int h = col >> 6, dh = col & 63;
    Vt[(((size_t)(b * Hh + h) * DH + dh) << 11) + s] = V[idx];
}

// ---------------------------------------------------------------------------
// LayerNorm matching the reference: y = g*(x-mean)/(std+eps) + b, std = sqrt(var)
// One block per row of 1024. Output bf16 (GEMM input).
// ---------------------------------------------------------------------------
__global__ __launch_bounds__(256) void ln_kernel(const float* __restrict__ X,
                                                 const float* __restrict__ g,
                                                 const float* __restrict__ bt,
                                                 __bf16* __restrict__ Y) {
    int row = blockIdx.x;
    const float* x = X + (size_t)row * Dm;
    float v[4], s = 0.f, s2 = 0.f;
    for (int i = 0; i < 4; ++i) {
        v[i] = x[threadIdx.x + i * 256];
        s += v[i];
        s2 += v[i] * v[i];
    }
    // 16-lane DPP reduction, then one cross-half hop.
    s  = redsum16(s);
    s2 = redsum16(s2);
    s  += __shfl_xor(s, 16, 32);
    s2 += __shfl_xor(s2, 16, 32);
    __shared__ float rs[8], rs2[8];
    int w = threadIdx.x >> 5, ln = threadIdx.x & 31;
    if (ln == 0) { rs[w] = s; rs2[w] = s2; }
    __syncthreads();
    if (w == 0) {
        float a = (ln < 8) ? rs[ln] : 0.f;
        float b2 = (ln < 8) ? rs2[ln] : 0.f;
        for (int i = 1; i < 8; i <<= 1) {
            a  += __shfl_xor(a, i, 32);
            b2 += __shfl_xor(b2, i, 32);
        }
        if (ln == 0) { rs[0] = a; rs2[0] = b2; }
    }
    __syncthreads();
    float mean = rs[0] * (1.f / (float)Dm);
    float var  = rs2[0] * (1.f / (float)Dm) - mean * mean;
    float inv  = 1.f / (sqrtf(fmaxf(var, 0.f)) + 1e-5f);
    for (int i = 0; i < 4; ++i) {
        int c = threadIdx.x + i * 256;
        float y = g[c] * ((v[i] - mean) * inv) + bt[c];
        Y[(size_t)row * Dm + c] = (__bf16)y;
    }
}

// ---------------------------------------------------------------------------
// Tiled bf16 WMMA GEMM: C[M,N](f32 acc) = A[M,K] * BT[N,K]^T  (+ epilogue)
//   MODE 0: store bf16 scaled by oscale
//   MODE 1: + residual(f32), store f32
//   MODE 2: + bias, exact GELU, store bf16
//   MODE 3: + bias + residual, store f32
// Block tile 128x128, 8 waves (2x4), wave tile 64x32, K step 32.
// ---------------------------------------------------------------------------
template <int MODE>
__global__ __launch_bounds__(256) void gemm_bf16(const __bf16* __restrict__ A,
                                                 const __bf16* __restrict__ BT,
                                                 const float* __restrict__ bias,
                                                 const float* __restrict__ res,
                                                 float* __restrict__ outF,
                                                 __bf16* __restrict__ outB,
                                                 int M, int N, int K, float oscale) {
    __shared__ __bf16 lA[128][40];
    __shared__ __bf16 lB[128][40];

    const int tid = threadIdx.x;
    const int wid = tid >> 5, lane = tid & 31, li = lane & 15;
    const bool lo = lane < 16;
    const int wm = wid >> 2, wn = wid & 3;
    const int m0 = blockIdx.y * 128, n0 = blockIdx.x * 128;

    v8f acc[4][2];
    for (int i = 0; i < 4; ++i)
        for (int j = 0; j < 2; ++j) acc[i][j] = vzero8();

    const int sr = tid >> 1;            // staging row 0..127
    const int sh = (tid & 1) * 16;      // staging half-row (16 bf16 = 32B)

    for (int k0 = 0; k0 < K; k0 += 32) {
        // Stage A tile [128 x 32] and BT tile [128 x 32] (contiguous in K).
        const uint4* gA = reinterpret_cast<const uint4*>(A + (size_t)(m0 + sr) * K + k0 + sh);
        uint4* sA = reinterpret_cast<uint4*>(&lA[sr][sh]);
        sA[0] = gA[0]; sA[1] = gA[1];
        const uint4* gB = reinterpret_cast<const uint4*>(BT + (size_t)(n0 + sr) * K + k0 + sh);
        uint4* sB = reinterpret_cast<uint4*>(&lB[sr][sh]);
        sB[0] = gB[0]; sB[1] = gB[1];
        __syncthreads();

        // A fragments (16x32): lane<16 -> K{0..7,16..23}; lane>=16 -> K{8..15,24..31}
        FragU af[4];
        for (int sm = 0; sm < 4; ++sm) {
            const uint4* p = reinterpret_cast<const uint4*>(&lA[wm * 64 + sm * 16 + li][0]);
            af[sm].q[0] = p[lo ? 0 : 1];
            af[sm].q[1] = p[lo ? 2 : 3];
        }
        // B fragments (32x16): lane<16 -> K{0..7,16..23}; lane>=16 -> K{8..15,24..31}
        FragU bfr[2];
        for (int sn = 0; sn < 2; ++sn) {
            const uint4* p = reinterpret_cast<const uint4*>(&lB[wn * 32 + sn * 16 + li][0]);
            bfr[sn].q[0] = p[lo ? 0 : 1];
            bfr[sn].q[1] = p[lo ? 2 : 3];
        }
        for (int sm = 0; sm < 4; ++sm)
            for (int sn = 0; sn < 2; ++sn)
                acc[sm][sn] = wmma_bf16(af[sm].v, bfr[sn].v, acc[sm][sn]);
        __syncthreads();
    }

    // Epilogue. D layout: VGPR e -> row (lo? e : 8+e), lane -> col li.
    for (int sm = 0; sm < 4; ++sm)
        for (int sn = 0; sn < 2; ++sn) {
            int ncol = n0 + wn * 32 + sn * 16 + li;
            for (int e = 0; e < 8; ++e) {
                int mrow = m0 + wm * 64 + sm * 16 + (lo ? e : 8 + e);
                size_t off = (size_t)mrow * N + ncol;
                float val = acc[sm][sn][e];
                if (MODE == 0) {
                    outB[off] = (__bf16)(val * oscale);
                } else if (MODE == 1) {
                    outF[off] = val + res[off];
                } else if (MODE == 2) {
                    val += bias[ncol];
                    val = 0.5f * val * (1.f + erff(val * 0.70710678118f));
                    outB[off] = (__bf16)val;
                } else {
                    val += bias[ncol];
                    outF[off] = val + res[off];
                }
            }
        }
}

// ---------------------------------------------------------------------------
// Causal flash attention. Grid: (S/64, H, B); 128 threads = 4 waves.
// Each wave owns 16 query rows; key blocks of 32; WMMA for QK^T and PV.
// Q (pre-scaled by 1/sqrt(DH)), K: [tok, D] bf16; Vt: [b,h,64,S] bf16;
// ctx: [tok, D] bf16.
// ---------------------------------------------------------------------------
__global__ __launch_bounds__(128) void attn_kernel(const __bf16* __restrict__ Qm,
                                                   const __bf16* __restrict__ Km,
                                                   const __bf16* __restrict__ Vt,
                                                   __bf16* __restrict__ ctx) {
    __shared__ __bf16 lP[4][16][40];   // per-wave P scratch (D-layout -> A-layout)
    const int tid = threadIdx.x, w = tid >> 5, lane = tid & 31, li = lane & 15;
    const bool lo = lane < 16;
    const int b = blockIdx.z, h = blockIdx.y;
    const int qbase = blockIdx.x * 64 + w * 16;
    const int q = qbase + li;

    // Q A-fragments, head-dim chunks c=0 (0..31), c=1 (32..63); loaded once.
    FragU qf[2];
    const __bf16* qp = Qm + ((size_t)(b * Sm + q) * Dm + h * DH);
    for (int c = 0; c < 2; ++c) {
        int k0 = 32 * c + (lo ? 0 : 8);
        qf[c].q[0] = *reinterpret_cast<const uint4*>(qp + k0);
        qf[c].q[1] = *reinterpret_cast<const uint4*>(qp + k0 + 16);
    }

    v8f m_i, l_i, O[4];
    for (int e = 0; e < 8; ++e) { m_i[e] = -3.0e38f; l_i[e] = 0.f; }
    for (int t = 0; t < 4; ++t) O[t] = vzero8();

    const int nkb = (qbase >> 5) + 1;       // causal: keys up to qbase+15
    for (int kb = 0; kb < nkb; ++kb) {
        // --- scores: S = Q K^T (two 16x16 N-subtiles, K=64 split as 2x32) ---
        v8f s[2];
        for (int kn = 0; kn < 2; ++kn) {
            v8f a = vzero8();
            int kr = kb * 32 + kn * 16 + li;
            for (int c = 0; c < 2; ++c) {
                const __bf16* kp = Km + ((size_t)(b * Sm + kr) * Dm + h * DH) +
                                   32 * c + (lo ? 0 : 8);
                FragU kf;
                kf.q[0] = *reinterpret_cast<const uint4*>(kp);
                kf.q[1] = *reinterpret_cast<const uint4*>(kp + 16);
                a = wmma_bf16(qf[c].v, kf.v, a);
            }
            s[kn] = a;
        }
        // causal mask (Q already carries the 1/sqrt(DH) scale)
        for (int kn = 0; kn < 2; ++kn)
            for (int e = 0; e < 8; ++e) {
                int key = kb * 32 + kn * 16 + li;
                int qr = qbase + (lo ? e : 8 + e);
                s[kn][e] = (key > qr) ? -3.0e38f : s[kn][e];
            }
        // --- online softmax (DPP row_ror butterflies, no LDS) ---
        v8f tmax;
        for (int e = 0; e < 8; ++e)
            tmax[e] = redmax16(fmaxf(s[0][e], s[1][e]));
        v8f mnew, scal, rsum;
        for (int e = 0; e < 8; ++e) {
            mnew[e] = fmaxf(m_i[e], tmax[e]);
            scal[e] = __expf(m_i[e] - mnew[e]);
        }
        for (int e = 0; e < 8; ++e) {
            float p0 = __expf(s[0][e] - mnew[e]);
            float p1 = __expf(s[1][e] - mnew[e]);
            s[0][e] = p0;
            s[1][e] = p1;
            rsum[e] = redsum16(p0 + p1);
        }
        for (int e = 0; e < 8; ++e) {
            l_i[e] = l_i[e] * scal[e] + rsum[e];
            m_i[e] = mnew[e];
        }
        for (int t = 0; t < 4; ++t)
            for (int e = 0; e < 8; ++e) O[t][e] *= scal[e];

        // --- P: D-layout -> A-layout via per-wave LDS ---
        for (int kn = 0; kn < 2; ++kn)
            for (int e = 0; e < 8; ++e) {
                int mr = lo ? e : 8 + e;
                lP[w][mr][kn * 16 + li] = (__bf16)s[kn][e];
            }
        asm volatile("s_wait_dscnt 0" ::: "memory");
        FragU pf;
        const uint4* pp = reinterpret_cast<const uint4*>(&lP[w][li][0]);
        pf.q[0] = pp[lo ? 0 : 1];
        pf.q[1] = pp[lo ? 2 : 3];

        // --- O += P * V (4 dh subtiles) ---
        for (int t = 0; t < 4; ++t) {
            const __bf16* vp = Vt + (((size_t)((b * Hh + h) * DH + t * 16 + li)) << 11) +
                               kb * 32 + (lo ? 0 : 8);
            FragU vf;
            vf.q[0] = *reinterpret_cast<const uint4*>(vp);
            vf.q[1] = *reinterpret_cast<const uint4*>(vp + 16);
            O[t] = wmma_bf16(pf.v, vf.v, O[t]);
        }
    }

    // normalize + store ctx (bf16)
    for (int t = 0; t < 4; ++t)
        for (int e = 0; e < 8; ++e) {
            int mr = lo ? e : 8 + e;
            float v = O[t][e] / l_i[e];
            ctx[((size_t)(b * Sm + qbase + mr) * Dm) + h * DH + t * 16 + li] = (__bf16)v;
        }
}

// ---------------------------------------------------------------------------
// Host orchestration
// ---------------------------------------------------------------------------
extern "C" void kernel_launch(void* const* d_in, const int* in_sizes, int n_in,
                              void* d_out, int out_size, void* d_ws, size_t ws_size,
                              hipStream_t stream) {
    const float* x   = (const float*)d_in[0];
    const float* Wq  = (const float*)d_in[1];
    const float* Wk  = (const float*)d_in[2];
    const float* Wv  = (const float*)d_in[3];
    const float* Wo  = (const float*)d_in[4];
    const float* g1  = (const float*)d_in[5];
    const float* b1  = (const float*)d_in[6];
    const float* g2  = (const float*)d_in[7];
    const float* b2  = (const float*)d_in[8];
    const float* W1  = (const float*)d_in[9];
    const float* bf1 = (const float*)d_in[10];
    const float* W2  = (const float*)d_in[11];
    const float* bf2 = (const float*)d_in[12];
    float* out = (float*)d_out;

    char* ws = (char*)d_ws;
    const size_t MB = 1024 * 1024;
    __bf16* WqT = (__bf16*)(ws + 0 * MB);    // 2 MB
    __bf16* WkT = (__bf16*)(ws + 2 * MB);    // 2 MB
    __bf16* WvT = (__bf16*)(ws + 4 * MB);    // 2 MB
    __bf16* WoT = (__bf16*)(ws + 6 * MB);    // 2 MB
    __bf16* W1T = (__bf16*)(ws + 8 * MB);    // 8 MB
    __bf16* W2T = (__bf16*)(ws + 16 * MB);   // 8 MB
    __bf16* h1  = (__bf16*)(ws + 24 * MB);   // 8 MB
    __bf16* Qb  = (__bf16*)(ws + 32 * MB);   // 8 MB
    __bf16* Kb  = (__bf16*)(ws + 40 * MB);   // 8 MB
    __bf16* Vb  = (__bf16*)(ws + 48 * MB);   // 8 MB
    __bf16* Vt  = (__bf16*)(ws + 56 * MB);   // 8 MB
    __bf16* ctx = (__bf16*)(ws + 64 * MB);   // 8 MB
    float*  x1  = (float*)(ws + 72 * MB);    // 16 MB
    __bf16* h2  = (__bf16*)(ws + 88 * MB);   // 8 MB
    __bf16* f1  = (__bf16*)(ws + 96 * MB);   // 32 MB

    // 1) weight transposes (f32 -> bf16, K-contiguous for B fragments)
    wt_kernel<<<(1024 * 1024) / 256, 256, 0, stream>>>(Wq, WqT, Dm, Dm);
    wt_kernel<<<(1024 * 1024) / 256, 256, 0, stream>>>(Wk, WkT, Dm, Dm);
    wt_kernel<<<(1024 * 1024) / 256, 256, 0, stream>>>(Wv, WvT, Dm, Dm);
    wt_kernel<<<(1024 * 1024) / 256, 256, 0, stream>>>(Wo, WoT, Dm, Dm);
    wt_kernel<<<(1024 * 4096) / 256, 256, 0, stream>>>(W1, W1T, Dm, FF);
    wt_kernel<<<(4096 * 1024) / 256, 256, 0, stream>>>(W2, W2T, FF, Dm);

    // 2) LN1 -> bf16
    ln_kernel<<<NTOK, 256, 0, stream>>>(x, g1, b1, h1);

    // 3) QKV projections (M=4096, N=1024, K=1024); Q pre-scaled by 1/sqrt(DH)
    dim3 gQ(Dm / 128, NTOK / 128);
    gemm_bf16<0><<<gQ, 256, 0, stream>>>(h1, WqT, nullptr, nullptr, nullptr, Qb,
                                         NTOK, Dm, Dm, 0.125f);
    gemm_bf16<0><<<gQ, 256, 0, stream>>>(h1, WkT, nullptr, nullptr, nullptr, Kb,
                                         NTOK, Dm, Dm, 1.0f);
    gemm_bf16<0><<<gQ, 256, 0, stream>>>(h1, WvT, nullptr, nullptr, nullptr, Vb,
                                         NTOK, Dm, Dm, 1.0f);

    // 4) V -> Vt (keys contiguous for PV B fragments)
    vt_kernel<<<(NTOK * Dm) / 256, 256, 0, stream>>>(Vb, Vt);

    // 5) causal flash attention -> ctx bf16
    attn_kernel<<<dim3(Sm / 64, Hh, 2), 128, 0, stream>>>(Qb, Kb, Vt, ctx);

    // 6) output projection + residual -> x1 f32
    gemm_bf16<1><<<gQ, 256, 0, stream>>>(ctx, WoT, nullptr, x, x1, nullptr,
                                         NTOK, Dm, Dm, 1.0f);

    // 7) LN2 -> bf16
    ln_kernel<<<NTOK, 256, 0, stream>>>(x1, g2, b2, h2);

    // 8) FFN1 + bias + exact GELU -> f1 bf16 (M=4096, N=4096, K=1024)
    gemm_bf16<2><<<dim3(FF / 128, NTOK / 128), 256, 0, stream>>>(h2, W1T, bf1, nullptr,
                                                                 nullptr, f1, NTOK, FF, Dm,
                                                                 1.0f);

    // 9) FFN2 + bias + residual -> out f32 (M=4096, N=1024, K=4096)
    gemm_bf16<3><<<gQ, 256, 0, stream>>>(f1, W2T, bf2, x1, out, nullptr,
                                         NTOK, Dm, FF, 1.0f);
}
